// SimpleRNN_31190052504305
// MI455X (gfx1250) — matrix operand order repaired and checked by reading, not compiled, not run on hardware
//
#include <hip/hip_runtime.h>

// ---------------------------------------------------------------------------
// Fused masked tanh-RNN scan for MI455X (gfx1250, wave32, WMMA).
//   h_t = tanh(x_t @ Wx^T + h_{t-1} @ Wh^T + bx + bh)   (per-row length mask)
//   out = h_T @ Wout^T + bout
// 16 workgroups (16 batch rows each) x 8 waves; wave w owns hidden columns
// [16w,16w+16) as a resident 16x16 f32 accumulator. Per step:
//   6 x v_wmma_f32_16x16x32_bf16 in 3 independent accumulation chains,
//   loop-invariant B operands (Wx/Wh tiles) resident in 48 VGPRs,
//   branch-free hardware tanh (v_exp_f32 + v_rcp_f32),
//   1 barrier/step with double-buffered LDS staging,
//   next x_t prefetched with nontemporal global_load_b128.
// ---------------------------------------------------------------------------

typedef __bf16 bf16;
typedef __attribute__((ext_vector_type(16))) __bf16 v16bf;
typedef __attribute__((ext_vector_type(8)))  __bf16 v8bf;
typedef __attribute__((ext_vector_type(8)))  float  v8f;
typedef __attribute__((ext_vector_type(4)))  float  v4f;

#define BB   256
#define TT   2048
#define DD   64
#define HID  128
#define NCLS 2

#define ROWS 16          // batch rows per workgroup
#define WXP  (DD + 8)    // padded row (bf16) for x / Wx tiles  -> 144 B rows
#define WHP  (HID + 8)   // padded row (bf16) for h / Wh tiles  -> 272 B rows

// A operand (16x32 bf16): lane = row (&15); kbase=(lane>>4)*16.
// Lane holds 16 consecutive K values kbase..kbase+15 of its row.
static __device__ __forceinline__ v16bf load_a(const bf16* row, int koff)
{
    v8bf lo = *(const v8bf*)(row + koff);
    v8bf hi = *(const v8bf*)(row + koff + 8);
    return __builtin_shufflevector(lo, hi,
        0,1,2,3,4,5,6,7,8,9,10,11,12,13,14,15);
}

// B operand (32x16 bf16): lane = column (&15); kbase=(lane>>4)*16.
// Lane holds 16 consecutive K values of its column.
static __device__ __forceinline__ v16bf load_b(const bf16* row, int koff)
{
    v8bf lo = *(const v8bf*)(row + koff);
    v8bf hi = *(const v8bf*)(row + koff + 8);
    return __builtin_shufflevector(lo, hi,
        0,1,2,3,4,5,6,7,8,9,10,11,12,13,14,15);
}

// Branch-free tanh: 1 - 2/(e^{2x}+1); v_exp_f32 is 2^x on AMD.
// x -> +inf: exp2 -> inf, rcp -> 0, result 1.  x -> -inf: exp2 -> 0, result -1.
static __device__ __forceinline__ float fast_tanh(float x)
{
    const float e = __builtin_amdgcn_exp2f(x * 2.88539008177793f); // 2*log2(e)
    const float r = __builtin_amdgcn_rcpf(e + 1.0f);
    return __builtin_fmaf(-2.0f, r, 1.0f);
}

__global__ __launch_bounds__(256, 1)
void rnn_scan_fused(const float* __restrict__ x_pad,
                    const int*   __restrict__ lengths,
                    const float* __restrict__ Wx,
                    const float* __restrict__ bx,
                    const float* __restrict__ Wh,
                    const float* __restrict__ bh,
                    const float* __restrict__ Wout,
                    const float* __restrict__ bout,
                    float*       __restrict__ out)
{
    __shared__ bf16  s_wx[HID][WXP];        // Wx row-major [n][d], bf16
    __shared__ bf16  s_wh[HID][WHP];        // Wh row-major [n][k], bf16
    __shared__ bf16  s_x[2][ROWS][WXP];     // x_t tile, double buffered
    __shared__ bf16  s_h[2][ROWS][WHP];     // h_t tile (bf16), double buffered
    __shared__ float s_bias[HID];           // bx + bh
    __shared__ float s_hf[ROWS][HID + 4];   // final f32 h for the head

    const int tid  = threadIdx.x;
    const int wave = tid >> 5;
    const int lane = tid & 31;
    const int b0   = blockIdx.x * ROWS;

    // ---- stage weights to LDS as bf16 (once) ------------------------------
    for (int i = tid; i < HID * DD;  i += 256) s_wx[i >> 6][i & (DD - 1)]  = (bf16)Wx[i];
    for (int i = tid; i < HID * HID; i += 256) s_wh[i >> 7][i & (HID - 1)] = (bf16)Wh[i];
    if (tid < HID) s_bias[tid] = bx[tid] + bh[tid];

    const int n0   = wave * 16;
    const int ncol = lane & 15;
    const int arow = lane & 15;
    const int akb  = (lane >> 4) * 16;
    const int bkb  = (lane >> 4) * 16;

    int lenv[8];
#pragma unroll
    for (int i = 0; i < 8; ++i)
        lenv[i] = lengths[b0 + ((lane < 16) ? i : i + 8)];

    // ---- x streaming (each thread owns one float4 of the 16x64 tile) ------
    const int xr_row = tid >> 4;          // 0..15
    const int xr_d0  = (tid & 15) * 4;    // 0..60 step 4
    const float* xptr = x_pad + ((size_t)(b0 + xr_row) * TT) * DD + xr_d0;

    v4f xr = __builtin_nontemporal_load((const v4f*)xptr);   // t = 0

    __syncthreads();
    const float bias_n = s_bias[n0 + ncol];

    // ---- loop-invariant B operands: Wx (2 tiles) + Wh (4 tiles) in VGPRs --
    v16bf bwx[2], bwh[4];
#pragma unroll
    for (int kc = 0; kc < 2; ++kc)
        bwx[kc] = load_b(&s_wx[n0 + ncol][0], kc * 32 + bkb);
#pragma unroll
    for (int kc = 0; kc < 4; ++kc)
        bwh[kc] = load_b(&s_wh[n0 + ncol][0], kc * 32 + bkb);

    v8f h = {};   // resident f32 hidden state (wave's 16x16 C tile)

    for (int t = 0; t < TT; ++t) {
        const int buf = t & 1;

        // publish x_t (bf16) and h_t (bf16) into this step's buffer
        s_x[buf][xr_row][xr_d0 + 0] = (bf16)xr[0];
        s_x[buf][xr_row][xr_d0 + 1] = (bf16)xr[1];
        s_x[buf][xr_row][xr_d0 + 2] = (bf16)xr[2];
        s_x[buf][xr_row][xr_d0 + 3] = (bf16)xr[3];
#pragma unroll
        for (int i = 0; i < 8; ++i) {
            const int m = (lane < 16) ? i : i + 8;
            s_h[buf][m][n0 + ncol] = (bf16)h[i];
        }
        __syncthreads();

        // prefetch next x_t while the WMMAs run (streamed, non-temporal)
        if (t + 1 < TT)
            xr = __builtin_nontemporal_load(
                     (const v4f*)(xptr + (size_t)(t + 1) * DD));

        // gather all A operands for this step (12 ds_load_b128)
        v16bf ax[2], ah[4];
#pragma unroll
        for (int kc = 0; kc < 2; ++kc)
            ax[kc] = load_a(&s_x[buf][arow][0], kc * 32 + akb);
#pragma unroll
        for (int kc = 0; kc < 4; ++kc)
            ah[kc] = load_a(&s_h[buf][arow][0], kc * 32 + akb);

        // 3 independent accumulation chains (2 WMMAs deep each)
        v8f c0, c1 = {}, c2 = {};
#pragma unroll
        for (int i = 0; i < 8; ++i) c0[i] = bias_n;   // bias folded into acc

        c0 = __builtin_amdgcn_wmma_f32_16x16x32_bf16(false, ax[0], false, bwx[0], (short)0, c0, false, false);
        c1 = __builtin_amdgcn_wmma_f32_16x16x32_bf16(false, ah[0], false, bwh[0], (short)0, c1, false, false);
        c2 = __builtin_amdgcn_wmma_f32_16x16x32_bf16(false, ah[2], false, bwh[2], (short)0, c2, false, false);
        c0 = __builtin_amdgcn_wmma_f32_16x16x32_bf16(false, ax[1], false, bwx[1], (short)0, c0, false, false);
        c1 = __builtin_amdgcn_wmma_f32_16x16x32_bf16(false, ah[1], false, bwh[1], (short)0, c1, false, false);
        c2 = __builtin_amdgcn_wmma_f32_16x16x32_bf16(false, ah[3], false, bwh[3], (short)0, c2, false, false);

        // tanh + per-row length mask (frozen rows keep exact f32 h)
#pragma unroll
        for (int i = 0; i < 8; ++i) {
            const float hn = fast_tanh(c0[i] + c1[i] + c2[i]);
            h[i] = (t < lenv[i]) ? hn : h[i];
        }
        // next iteration writes the other buffer; its barrier also fences
        // this iteration's LDS reads, so one barrier per step suffices.
    }

    // ---- head: out = h_T @ Wout^T + bout ----------------------------------
#pragma unroll
    for (int i = 0; i < 8; ++i) {
        const int m = (lane < 16) ? i : i + 8;
        s_hf[m][n0 + ncol] = h[i];
    }
    __syncthreads();

    if (tid < ROWS * NCLS) {
        const int r  = tid >> 1;
        const int cc = tid & 1;
        float s = bout[cc];
        for (int k = 0; k < HID; ++k)
            s += s_hf[r][k] * Wout[cc * HID + k];
        out[(size_t)(b0 + r) * NCLS + cc] = s;
    }
}

extern "C" void kernel_launch(void* const* d_in, const int* in_sizes, int n_in,
                              void* d_out, int out_size, void* d_ws, size_t ws_size,
                              hipStream_t stream)
{
    const float* x_pad   = (const float*)d_in[0];
    const int*   lengths = (const int*)  d_in[1];
    const float* Wx      = (const float*)d_in[2];
    const float* bx      = (const float*)d_in[3];
    const float* Wh      = (const float*)d_in[4];
    const float* bh      = (const float*)d_in[5];
    const float* Wout    = (const float*)d_in[6];
    const float* bout    = (const float*)d_in[7];
    float*       out     = (float*)d_out;

    (void)in_sizes; (void)n_in; (void)out_size; (void)d_ws; (void)ws_size;

    rnn_scan_fused<<<dim3(BB / ROWS), dim3(256), 0, stream>>>(
        x_pad, lengths, Wx, bx, Wh, bh, Wout, bout, out);
}
